// EncoderLayer_17411797418401
// MI455X (gfx1250) — matrix operand order repaired and checked by reading, not compiled, not run on hardware
//
#include <hip/hip_runtime.h>
#include <cstdint>
#include <cstddef>

// ---------------------------------------------------------------------------
// Problem constants (match reference)
// ---------------------------------------------------------------------------
static constexpr int Bq  = 2;
static constexpr int SEQ = 2048;
static constexpr int DM  = 1024;   // model dim
static constexpr int NH  = 16;     // heads
static constexpr int DH  = 64;     // head dim
static constexpr int FF  = 4096;   // ffn dim
static constexpr int MR  = Bq * SEQ;  // 4096 rows of activations

// ---------------------------------------------------------------------------
// WMMA vector types (CDNA5 / gfx1250, wave32)
// ---------------------------------------------------------------------------
typedef __attribute__((ext_vector_type(8)))  float  v8f;
typedef __attribute__((ext_vector_type(16))) __bf16 v16bf;
typedef __attribute__((ext_vector_type(8)))  __bf16 v8bf;

union V16 { v16bf v; v8bf h[2]; };
union U8  { uint4 q; unsigned short s[8]; };

__device__ __forceinline__ unsigned short f2bf(float f) {
    unsigned u = __float_as_uint(f);
    u += 0x7FFFu + ((u >> 16) & 1u);          // round-to-nearest-even
    return (unsigned short)(u >> 16);
}

__device__ __forceinline__ v8f wmma_bf16(const v16bf& a, const v16bf& b, const v8f& c) {
    return __builtin_amdgcn_wmma_f32_16x16x32_bf16(
        /*neg_a=*/false, a, /*neg_b=*/false, b,
        /*c_mod=*/(short)0, c, /*reuse_a=*/false, /*reuse_b=*/false);
}

// ---------------------------------------------------------------------------
// fp32 -> bf16 conversion (weights)
// ---------------------------------------------------------------------------
__global__ void cvt_bf16_kernel(const float* __restrict__ in,
                                unsigned short* __restrict__ out, int n) {
    int i = blockIdx.x * blockDim.x + threadIdx.x;
    if (i < n) out[i] = f2bf(in[i]);
}

// ---------------------------------------------------------------------------
// LayerNorm (one 256-thread block per row of 1024), out bf16
// matches reference: (x-mean)/(std+eps), std = sqrt(mean((x-mean)^2))
// ---------------------------------------------------------------------------
__global__ __launch_bounds__(256)
void ln_kernel(const float* __restrict__ x, const float* __restrict__ g,
               const float* __restrict__ b, unsigned short* __restrict__ out) {
    const int row = blockIdx.x;
    const int tid = threadIdx.x;
    const float* xr = x + (size_t)row * DM;
    float vals[4], s1 = 0.f, s2 = 0.f;
#pragma unroll
    for (int i = 0; i < 4; ++i) {
        float v = xr[tid + i * 256];
        vals[i] = v; s1 += v; s2 += v * v;
    }
    __shared__ float r1[256], r2[256];
    r1[tid] = s1; r2[tid] = s2;
    __syncthreads();
    for (int off = 128; off > 0; off >>= 1) {
        if (tid < off) { r1[tid] += r1[tid + off]; r2[tid] += r2[tid + off]; }
        __syncthreads();
    }
    const float mean = r1[0] * (1.f / DM);
    const float var  = fmaxf(r2[0] * (1.f / DM) - mean * mean, 0.f);
    const float inv  = 1.f / (sqrtf(var) + 1e-6f);
#pragma unroll
    for (int i = 0; i < 4; ++i) {
        int c = tid + i * 256;
        out[(size_t)row * DM + c] = f2bf(g[c] * (vals[i] - mean) * inv + b[c]);
    }
}

// ---------------------------------------------------------------------------
// Tiled bf16 WMMA GEMM:  C[M,N] = A[M,K] * W[N,K]^T   (+ epilogue by MODE)
//   MODE 0: scatter bf16 to QKV layout [3][B][H][S][DH]
//   MODE 1: outF = resid + gemm                     (fp32)
//   MODE 2: outB = bf16( gelu(gemm + bias) )
//   MODE 3: outF = resid + gemm + bias              (fp32)
// Block: 256 threads = 8 waves; block tile 128x128; K step 32.
// Wave grid 2x4 -> each wave owns a 64x32 output tile (4x2 WMMA frags).
// ---------------------------------------------------------------------------
template <int MODE>
__global__ __launch_bounds__(256)
void gemm_bf16_kernel(const unsigned short* __restrict__ A,
                      const unsigned short* __restrict__ W,
                      const float* __restrict__ bias,
                      const float* __restrict__ resid,
                      float* __restrict__ outF,
                      unsigned short* __restrict__ outB,
                      int M, int N, int K) {
    __shared__ unsigned short As[128][40];   // 80 B row stride (16B aligned)
    __shared__ unsigned short Bt[32][136];   // transposed weights: [k][n], 272 B stride

    const int tid  = threadIdx.x;
    const int wave = tid >> 5, lane = tid & 31;
    const int hh = lane >> 4, ln = lane & 15;
    const int wm = wave >> 2, wn = wave & 3;          // 2 x 4 wave grid
    const int m0 = blockIdx.y * 128, n0 = blockIdx.x * 128;

    v8f acc[4][2];
#pragma unroll
    for (int i = 0; i < 4; ++i)
#pragma unroll
        for (int j = 0; j < 2; ++j)
#pragma unroll
            for (int v = 0; v < 8; ++v) acc[i][j][v] = 0.f;

    for (int kt = 0; kt < K; kt += 32) {
        __syncthreads();
        // stage A tile (128 x 32 bf16) : 512 16-byte chunks
#pragma unroll
        for (int it = 0; it < 2; ++it) {
            int c = tid + it * 256;
            int row = c >> 2, kc = (c & 3) * 8;
            uint4 d = *(const uint4*)(A + (size_t)(m0 + row) * K + kt + kc);
            *(uint4*)&As[row][kc] = d;
        }
        // stage W tile transposed (Bt[k][n])
#pragma unroll
        for (int it = 0; it < 2; ++it) {
            int c = tid + it * 256;
            int row = c >> 2, kc = (c & 3) * 8;
            U8 u; u.q = *(const uint4*)(W + (size_t)(n0 + row) * K + kt + kc);
#pragma unroll
            for (int j = 0; j < 8; ++j) Bt[kc + j][row] = u.s[j];
        }
        __syncthreads();

        v16bf af[4], bf[2];
#pragma unroll
        for (int i = 0; i < 4; ++i) {
            int r = wm * 64 + i * 16 + ln;           // A-frag: lane row = lane&15
            V16 u;
            u.h[0] = *(const v8bf*)&As[r][hh * 8];        // K = h*8 .. +8
            u.h[1] = *(const v8bf*)&As[r][16 + hh * 8];   // K = 16+h*8 .. +8
            af[i] = u.v;
        }
#pragma unroll
        for (int j = 0; j < 2; ++j) {
            int n = wn * 32 + j * 16;                // B-frag: lane&15 = column N, elements = K
            V16 u;
#pragma unroll
            for (int s = 0; s < 16; ++s)
                u.v[s] = __builtin_bit_cast(__bf16, Bt[(s & 7) + hh * 8 + (s >> 3) * 16][n + ln]);
            bf[j] = u.v;
        }
#pragma unroll
        for (int i = 0; i < 4; ++i)
#pragma unroll
            for (int j = 0; j < 2; ++j)
                acc[i][j] = wmma_bf16(af[i], bf[j], acc[i][j]);
    }

    // epilogue: C layout -> lane holds (M = v + 8*hh, N = lane&15) per frag
#pragma unroll
    for (int i = 0; i < 4; ++i)
#pragma unroll
        for (int j = 0; j < 2; ++j)
#pragma unroll
            for (int v = 0; v < 8; ++v) {
                int m = m0 + wm * 64 + i * 16 + v + 8 * hh;
                int n = n0 + wn * 32 + j * 16 + ln;
                float val = acc[i][j][v];
                if constexpr (MODE == 2 || MODE == 3) val += bias[n];
                if constexpr (MODE == 2)
                    val = 0.5f * val * (1.0f + erff(val * 0.70710678118654752f));
                if constexpr (MODE == 1 || MODE == 3)
                    val += resid[(size_t)m * N + n];
                if constexpr (MODE == 0) {
                    int t = n >> 10, r = n & 1023;
                    int head = r >> 6, dh = r & 63;
                    int bb = m >> 11, s = m & 2047;
                    size_t idx = ((((size_t)t * Bq + bb) * NH + head) * SEQ + s) * DH + dh;
                    outB[idx] = f2bf(val);
                } else if constexpr (MODE == 2) {
                    outB[(size_t)m * N + n] = f2bf(val);
                } else {
                    outF[(size_t)m * N + n] = val;
                }
            }
}

// ---------------------------------------------------------------------------
// Flash attention (causal), bf16 in / bf16 out, fp32 accum + online softmax.
// Grid: (S/64 q-tiles, B*H). Block: 128 threads = 4 waves, each wave owns
// 16 q-rows. K tiles of 64 streamed through LDS.
// ---------------------------------------------------------------------------
__global__ __launch_bounds__(128)
void attn_kernel(const unsigned short* __restrict__ qkv,
                 unsigned short* __restrict__ out) {
    const int qt = blockIdx.x;
    const int bh = blockIdx.y;
    const int b  = bh / NH, h = bh % NH;
    const size_t plane = (size_t)Bq * NH * SEQ * DH;     // per q/k/v block
    const unsigned short* qp = qkv + (size_t)bh * SEQ * DH;
    const unsigned short* kp = qp + plane;
    const unsigned short* vp = qp + 2 * plane;

    __shared__ unsigned short Kt[DH][64 + 8];        // [d][j]  (transposed K)
    __shared__ unsigned short Vt[64][DH + 8];        // [j][d]
    __shared__ unsigned short Pt[4][16][64 + 8];     // per-wave probs

    const int tid  = threadIdx.x;
    const int wave = tid >> 5, lane = tid & 31;
    const int hh = lane >> 4, ln = lane & 15;
    const int i0 = qt * 64;
    const int qrow = i0 + wave * 16;

    // Q fragments for this wave's 16 rows (DH=64 -> two K-steps of 32)
    v16bf qf[2];
#pragma unroll
    for (int ks = 0; ks < 2; ++ks) {
        const unsigned short* base = qp + (size_t)(qrow + ln) * DH + ks * 32;
        V16 u;
        u.h[0] = *(const v8bf*)(base + hh * 8);
        u.h[1] = *(const v8bf*)(base + 16 + hh * 8);
        qf[ks] = u.v;
    }

    float m_i[8], l_i[8];
    v8f o_acc[4];
#pragma unroll
    for (int v = 0; v < 8; ++v) { m_i[v] = -3.0e38f; l_i[v] = 0.f; }
#pragma unroll
    for (int nt = 0; nt < 4; ++nt)
#pragma unroll
        for (int v = 0; v < 8; ++v) o_acc[nt][v] = 0.f;

    for (int kt = 0; kt <= qt; ++kt) {
        const int j0 = kt * 64;
        __syncthreads();
        // stage K (transposed) and V tiles: 512 chunks of 8 bf16
        for (int c = tid; c < 512; c += 128) {
            int row = c >> 3, kc = (c & 7) * 8;
            U8 uk; uk.q = *(const uint4*)(kp + (size_t)(j0 + row) * DH + kc);
#pragma unroll
            for (int j = 0; j < 8; ++j) Kt[kc + j][row] = uk.s[j];
            uint4 uv = *(const uint4*)(vp + (size_t)(j0 + row) * DH + kc);
            *(uint4*)&Vt[row][kc] = uv;
        }
        __syncthreads();

        // scores = Q K^T  (16 x 64), fp32 accum
        v8f sc[4];
#pragma unroll
        for (int nt = 0; nt < 4; ++nt) {
#pragma unroll
            for (int v = 0; v < 8; ++v) sc[nt][v] = 0.f;
#pragma unroll
            for (int ks = 0; ks < 2; ++ks) {
                V16 u;
#pragma unroll
                for (int s = 0; s < 16; ++s)
                    u.v[s] = __builtin_bit_cast(__bf16, Kt[ks * 32 + (s & 7) + hh * 8 + (s >> 3) * 16][nt * 16 + ln]);
                sc[nt] = wmma_bf16(qf[ks], u.v, sc[nt]);
            }
        }
        // scale + causal mask
#pragma unroll
        for (int nt = 0; nt < 4; ++nt)
#pragma unroll
            for (int v = 0; v < 8; ++v) {
                float val = sc[nt][v] * 0.125f;      // DH^-0.5
                int j = j0 + nt * 16 + ln;
                int i = qrow + v + 8 * hh;
                sc[nt][v] = (j > i) ? -3.0e38f : val;
            }
        // online softmax (rows live across the 16 lanes of each half)
        float mnew[8];
#pragma unroll
        for (int v = 0; v < 8; ++v) {
            float x = fmaxf(fmaxf(sc[0][v], sc[1][v]), fmaxf(sc[2][v], sc[3][v]));
#pragma unroll
            for (int msk = 1; msk < 16; msk <<= 1)
                x = fmaxf(x, __shfl_xor(x, msk, 32));
            mnew[v] = fmaxf(m_i[v], x);
        }
#pragma unroll
        for (int v = 0; v < 8; ++v) {
            float corr = expf(m_i[v] - mnew[v]);
            l_i[v] *= corr;
#pragma unroll
            for (int nt = 0; nt < 4; ++nt) o_acc[nt][v] *= corr;
        }
#pragma unroll
        for (int nt = 0; nt < 4; ++nt)
#pragma unroll
            for (int v = 0; v < 8; ++v)
                sc[nt][v] = expf(sc[nt][v] - mnew[v]);
#pragma unroll
        for (int v = 0; v < 8; ++v) {
            float rs = sc[0][v] + sc[1][v] + sc[2][v] + sc[3][v];
#pragma unroll
            for (int msk = 1; msk < 16; msk <<= 1)
                rs += __shfl_xor(rs, msk, 32);
            l_i[v] += rs;
            m_i[v] = mnew[v];
        }
        // reshape P (C layout) -> A-frag layout via per-wave LDS
#pragma unroll
        for (int nt = 0; nt < 4; ++nt)
#pragma unroll
            for (int v = 0; v < 8; ++v)
                Pt[wave][v + 8 * hh][nt * 16 + ln] = f2bf(sc[nt][v]);
        __syncthreads();
        // O += P V   (contraction over 64 key positions = two 32-steps)
#pragma unroll
        for (int ks = 0; ks < 2; ++ks) {
            V16 up;
            up.h[0] = *(const v8bf*)&Pt[wave][ln][ks * 32 + hh * 8];
            up.h[1] = *(const v8bf*)&Pt[wave][ln][ks * 32 + 16 + hh * 8];
#pragma unroll
            for (int nt = 0; nt < 4; ++nt) {
                V16 uv;
#pragma unroll
                for (int s = 0; s < 16; ++s)
                    uv.v[s] = __builtin_bit_cast(__bf16, Vt[ks * 32 + (s & 7) + hh * 8 + (s >> 3) * 16][nt * 16 + ln]);
                o_acc[nt] = wmma_bf16(up.v, uv.v, o_acc[nt]);
            }
        }
    }

    // write attnout[b][s][h*64 + d] as bf16 (row-major [M, DM] for next GEMM)
#pragma unroll
    for (int nt = 0; nt < 4; ++nt)
#pragma unroll
        for (int v = 0; v < 8; ++v) {
            int s = qrow + v + 8 * hh;
            int d = nt * 16 + ln;
            float o = o_acc[nt][v] / l_i[v];
            out[((size_t)(b * SEQ + s)) * DM + h * DH + d] = f2bf(o);
        }
}

// ---------------------------------------------------------------------------
// Host-side launcher
// ---------------------------------------------------------------------------
extern "C" void kernel_launch(void* const* d_in, const int* in_sizes, int n_in,
                              void* d_out, int out_size, void* d_ws, size_t ws_size,
                              hipStream_t stream) {
    (void)in_sizes; (void)n_in; (void)out_size; (void)ws_size;
    const float* x      = (const float*)d_in[0];
    // d_in[1] = attention_mask (causal, applied analytically) -- ignored
    const float* Wqkv   = (const float*)d_in[2];
    const float* Wout   = (const float*)d_in[3];
    const float* W1     = (const float*)d_in[4];
    const float* b1     = (const float*)d_in[5];
    const float* W2     = (const float*)d_in[6];
    const float* b2     = (const float*)d_in[7];
    const float* gamma1 = (const float*)d_in[8];
    const float* beta1  = (const float*)d_in[9];
    const float* gamma2 = (const float*)d_in[10];
    const float* beta2  = (const float*)d_in[11];
    float* outp = (float*)d_out;

    // workspace carve-up (256 B aligned)
    char* ws = (char*)d_ws;
    size_t off = 0;
    auto carve = [&](size_t bytes) -> char* {
        char* p = ws + off;
        off += (bytes + 255) & ~(size_t)255;
        return p;
    };
    unsigned short* wqkv_bf = (unsigned short*)carve((size_t)3 * DM * DM * 2);
    unsigned short* wout_bf = (unsigned short*)carve((size_t)DM * DM * 2);
    unsigned short* w1_bf   = (unsigned short*)carve((size_t)FF * DM * 2);
    unsigned short* w2_bf   = (unsigned short*)carve((size_t)DM * FF * 2);
    unsigned short* hA      = (unsigned short*)carve((size_t)MR * DM * 2);
    unsigned short* qkv     = (unsigned short*)carve((size_t)3 * MR * DM * 2);
    unsigned short* attnout = (unsigned short*)carve((size_t)MR * DM * 2);
    float*          x2      = (float*)carve((size_t)MR * DM * 4);
    unsigned short* h2      = (unsigned short*)carve((size_t)MR * DM * 2);
    unsigned short* act     = (unsigned short*)carve((size_t)MR * FF * 2);

    // 1) weights -> bf16
    auto cvt = [&](const float* src, unsigned short* dst, int n) {
        cvt_bf16_kernel<<<(n + 255) / 256, 256, 0, stream>>>(src, dst, n);
    };
    cvt(Wqkv, wqkv_bf, 3 * DM * DM);
    cvt(Wout, wout_bf, DM * DM);
    cvt(W1,   w1_bf,   FF * DM);
    cvt(W2,   w2_bf,   DM * FF);

    // 2) LN1
    ln_kernel<<<MR, 256, 0, stream>>>(x, gamma1, beta1, hA);

    // 3) QKV projection (scatter to [3][B][H][S][DH])
    gemm_bf16_kernel<0><<<dim3((3 * DM) / 128, MR / 128), 256, 0, stream>>>(
        hA, wqkv_bf, nullptr, nullptr, nullptr, qkv, MR, 3 * DM, DM);

    // 4) causal flash attention
    attn_kernel<<<dim3(SEQ / 64, Bq * NH), 128, 0, stream>>>(qkv, attnout);

    // 5) Wout projection + residual -> x2 (fp32)
    gemm_bf16_kernel<1><<<dim3(DM / 128, MR / 128), 256, 0, stream>>>(
        attnout, wout_bf, nullptr, x, x2, nullptr, MR, DM, DM);

    // 6) LN2
    ln_kernel<<<MR, 256, 0, stream>>>(x2, gamma2, beta2, h2);

    // 7) FFN up + bias + exact GELU (bf16)
    gemm_bf16_kernel<2><<<dim3(FF / 128, MR / 128), 256, 0, stream>>>(
        h2, w1_bf, b1, nullptr, nullptr, act, MR, FF, DM);

    // 8) FFN down + bias + residual -> d_out (fp32)
    gemm_bf16_kernel<3><<<dim3(DM / 128, MR / 128), 256, 0, stream>>>(
        act, w2_bf, b2, x2, outp, nullptr, MR, DM, FF);
}
